// SinkhornOTLoss_71932112273864
// MI455X (gfx1250) — compile-verified, matrix operand order checked by reading, not run on hardware
//
#include <hip/hip_runtime.h>
#include <stdint.h>

// ---------------------------------------------------------------------------
// Sinkhorn OT loss on MI455X (gfx1250), persistent-kernel formulation.
//   T=256, V=30000.  K = exp(-20*cost) is L2-resident (~92 MB << 192 MB L2).
//   K stored three ways (written once by init):
//     kpair : packed {bf16 hi, bf16 lo} per dword  -> s-phase / loss (f32 dot)
//     kh/kl : separate row-major bf16 planes       -> WMMA A operands loaded
//             directly as 2x global_load_b128 (no register repacking)
//   K*v uses v_wmma_f32_16x16x32_bf16 x3 (bf16-split Ah*vh+Al*vh+Ah*vl =>
//   ~f32 accuracy) with THREE independent accumulator chains for WMMA ILP;
//   vh/vl are split once per column and served from LDS as b128 loads.
//   One grid barrier per Sinkhorn iteration; all reductions use exclusive
//   ownership + fixed-order sums (no float atomics) => deterministic.
// ---------------------------------------------------------------------------

#define T_DIM   256
#define V_DIM   30000
#define S_ALPHA 20.0f
#define W_LOSS  250.0f
#define MAX_IT  500
#define STOPTHR 0.005f
#define EPS_F   1e-16f

#define NBLK    64          // persistent blocks (must all be co-resident)
#define TPB     512         // 16 waves of 32
#define COLS_PB 480         // 64*480 = 30720 >= 30000, multiple of 32
#define PAD_EL  64          // tail padding for over-read of last row

typedef __attribute__((ext_vector_type(16))) __bf16 v16bf;
typedef __attribute__((ext_vector_type(8)))  float  v8f;

union BfVec { unsigned short h[16]; uint4 q[2]; v16bf v; };

__device__ __forceinline__ unsigned short f32_to_bf16_rne(float x) {
  unsigned u = __float_as_uint(x);
  unsigned r = (u + 0x7FFFu + ((u >> 16) & 1u)) >> 16;
  return (unsigned short)r;
}
__device__ __forceinline__ float bfhi_f32(unsigned u) { return __uint_as_float(u & 0xFFFF0000u); }
__device__ __forceinline__ float bflo_f32(unsigned u) { return __uint_as_float(u << 16); }

// Monotonic-generation grid barrier (no counter reset; NBLK blocks).
__device__ __forceinline__ void grid_sync(unsigned* bar, unsigned gen) {
  __syncthreads();
  if (threadIdx.x == 0) {
    __threadfence();
    unsigned prev = __hip_atomic_fetch_add(bar + 0, 1u, __ATOMIC_ACQ_REL,
                                           __HIP_MEMORY_SCOPE_AGENT);
    if (prev == gen * (unsigned)NBLK - 1u) {
      __hip_atomic_store(bar + 1, gen, __ATOMIC_RELEASE, __HIP_MEMORY_SCOPE_AGENT);
    } else {
      while (__hip_atomic_load(bar + 1, __ATOMIC_ACQUIRE,
                               __HIP_MEMORY_SCOPE_AGENT) < gen)
        __builtin_amdgcn_s_sleep(2);
    }
    __threadfence();
  }
  __syncthreads();
}

// ---------------------------------------------------------------------------
// Init: K = exp(-alpha*cost); hi = bf16(K), lo = bf16(K - f32(hi)).
// Writes packed dwords + separate bf16 planes; zeroes pads and barrier state.
// ---------------------------------------------------------------------------
__global__ void sinkhorn_init(const float* __restrict__ cost,
                              unsigned* __restrict__ kpair,
                              unsigned short* __restrict__ kh,
                              unsigned short* __restrict__ kl,
                              unsigned* __restrict__ bar) {
  size_t idx = (size_t)blockIdx.x * blockDim.x + threadIdx.x;
  const size_t n = (size_t)T_DIM * V_DIM;
  if (idx < n) {
    float k  = expf(-S_ALPHA * cost[idx]);
    unsigned short hi = f32_to_bf16_rne(k);
    float res = k - __uint_as_float((unsigned)hi << 16);
    unsigned short lo = f32_to_bf16_rne(res);
    kpair[idx] = ((unsigned)hi << 16) | (unsigned)lo;
    kh[idx] = hi;
    kl[idx] = lo;
  } else if (idx < n + PAD_EL) {
    kpair[idx] = 0u;
    kh[idx] = 0;
    kl[idx] = 0;
  }
  if (idx < 2) bar[idx] = 0u;
}

// ---------------------------------------------------------------------------
// Persistent Sinkhorn loop.
//   slab     : [2][NBLK][T_DIM] per-block K*v row partials (double buffered,
//              plain stores only: wave w owns rows 16w..16w+15 exclusively)
//   err_part : [16][NBLK] per-block error partials per 50-iter check
//   partials : [NBLK] per-block loss partials
// ---------------------------------------------------------------------------
__global__ __launch_bounds__(TPB) void sinkhorn_persist(
    const float* __restrict__ cost,
    const unsigned* __restrict__ kpair,
    const unsigned short* __restrict__ kh,
    const unsigned short* __restrict__ kl,
    float* __restrict__ slab,
    float* __restrict__ err_part,
    float* __restrict__ partials,
    unsigned* __restrict__ bar) {
  __shared__ float u_lds[T_DIM];
  __shared__ __align__(16) unsigned short vh_lds[COLS_PB];
  __shared__ __align__(16) unsigned short vl_lds[COLS_PB];
  __shared__ float red[TPB];

  const int tid = threadIdx.x;
  const int b   = blockIdx.x;
  const int j0  = b * COLS_PB;
  const int myj = j0 + tid;
  const bool jvalid = (tid < COLS_PB) && (myj < V_DIM);
  int colsHere = V_DIM - j0;
  if (colsHere < 0) colsHere = 0;
  if (colsHere > COLS_PB) colsHere = COLS_PB;
  const int nchunks = (colsHere + 31) / 32;

  const int wav  = tid >> 5;
  const int lane = tid & 31;
  const int hi2  = lane >> 4;        // lane half (0 or 1)
  const int mrow = lane & 15;        // A-matrix M row within tile
  const int koff = 8 * hi2;          // ISA 16-bit A layout: K offset per half
  const size_t rowbase = (size_t)(16 * wav + mrow) * V_DIM;
  const unsigned short* rowh = kh + rowbase;
  const unsigned short* rowl = kl + rowbase;

  const float A0 = 1.0f / (float)T_DIM;
  const float B0 = 1.0f / (float)V_DIM;

  float v_reg = 0.0f;
  unsigned gen = 0;
  bool stopped = false;

  for (int iter = 0; iter < MAX_IT; ++iter) {
    // ---- u = a / (sum_b slab_prev + eps) -------------------------------
    if (tid < T_DIM) {
      float ut;
      if (iter == 0) {
        ut = A0;
      } else {
        const float* sl = slab + (size_t)((iter - 1) & 1) * (NBLK * T_DIM);
        float acc = 0.f;
        #pragma unroll 8
        for (int bb = 0; bb < NBLK; ++bb)
          acc += __hip_atomic_load(sl + bb * T_DIM + tid, __ATOMIC_RELAXED,
                                   __HIP_MEMORY_SCOPE_AGENT);
        ut = A0 / (acc + EPS_F);
      }
      u_lds[tid] = ut;
    }
    __syncthreads();

    // ---- s_j = (K^T u)_j  (lane = column, coalesced dwords, f32 hi+lo) -
    float sj = 0.f;
    if (jvalid) {
      const unsigned* colp = kpair + myj;
      #pragma unroll 8
      for (int t = 0; t < T_DIM; ++t) {
        unsigned kp = colp[t * V_DIM];
        float ut = u_lds[t];
        sj = fmaf(bfhi_f32(kp), ut, sj);
        sj = fmaf(bflo_f32(kp), ut, sj);
      }
    }

    // ---- convergence check (cnt % 50 == 0): OLD v vs K^T u_new ---------
    if (iter >= 50 && (iter % 50) == 0) {
      const int m = iter / 50 - 1;
      red[tid] = jvalid ? fabsf(v_reg * sj - B0) : 0.f;
      __syncthreads();
      for (int s2 = TPB / 2; s2 > 0; s2 >>= 1) {
        if (tid < s2) red[tid] += red[tid + s2];
        __syncthreads();
      }
      if (tid == 0)
        __hip_atomic_store(err_part + m * NBLK + b, red[0], __ATOMIC_RELAXED,
                           __HIP_MEMORY_SCOPE_AGENT);
      grid_sync(bar, ++gen);
      float errsum = 0.f;
      for (int bb = 0; bb < NBLK; ++bb)
        errsum += __hip_atomic_load(err_part + m * NBLK + bb, __ATOMIC_RELAXED,
                                    __HIP_MEMORY_SCOPE_AGENT);
      if (errsum <= STOPTHR) { stopped = true; break; }  // u_lds,v_reg = u^k,v^k
    }

    // ---- v = b / (s + eps); split v into bf16 hi/lo once per column ----
    if (jvalid) v_reg = B0 / (sj + EPS_F);
    if (tid < COLS_PB) {
      float bv = jvalid ? v_reg : 0.f;
      unsigned short hb = f32_to_bf16_rne(bv);
      vh_lds[tid] = hb;
      vl_lds[tid] = f32_to_bf16_rne(bv - __uint_as_float((unsigned)hb << 16));
    }
    __syncthreads();

    // ---- r = K v via WMMA (bf16-split: Ah*vh + Al*vh + Ah*vl) ----------
    // A operands load directly from bf16 planes (element order == memory
    // order per ISA 16-bit A layout); B operands load directly from LDS.
    // Three independent accumulator chains -> 3x WMMA pipeline ILP.
    v8f c0 = {}, c1 = {}, c2 = {};
    for (int ck = 0; ck < nchunks; ++ck) {
      const int jc = j0 + 32 * ck;
      BfVec ah, al, vh, vl;
      ah.q[0] = *(const uint4*)(rowh + jc + koff);
      ah.q[1] = *(const uint4*)(rowh + jc + 16 + koff);
      al.q[0] = *(const uint4*)(rowl + jc + koff);
      al.q[1] = *(const uint4*)(rowl + jc + 16 + koff);
      const int cb = 32 * ck;
      vh.q[0] = *(const uint4*)(vh_lds + cb + koff);
      vh.q[1] = *(const uint4*)(vh_lds + cb + 16 + koff);
      vl.q[0] = *(const uint4*)(vl_lds + cb + koff);
      vl.q[1] = *(const uint4*)(vl_lds + cb + 16 + koff);
      c0 = __builtin_amdgcn_wmma_f32_16x16x32_bf16(false, ah.v, false, vh.v,
                                                   (short)0, c0, false, false);
      c1 = __builtin_amdgcn_wmma_f32_16x16x32_bf16(false, al.v, false, vh.v,
                                                   (short)0, c1, false, false);
      c2 = __builtin_amdgcn_wmma_f32_16x16x32_bf16(false, ah.v, false, vl.v,
                                                   (short)0, c2, false, false);
    }
    // D column 0 (all N columns identical since B is broadcast across N):
    // VGPR g / lane-half hi2 holds row 16*wav + 8*hi2 + g.
    {
      float* sl = slab + (size_t)(iter & 1) * (NBLK * T_DIM) + b * T_DIM;
      if (mrow == 0) {
        #pragma unroll
        for (int g = 0; g < 8; ++g)
          __hip_atomic_store(sl + 16 * wav + 8 * hi2 + g,
                             c0[g] + c1[g] + c2[g],
                             __ATOMIC_RELAXED, __HIP_MEMORY_SCOPE_AGENT);
      }
    }
    grid_sync(bar, ++gen);
  }

  // ---- final u if loop hit the iteration cap ---------------------------
  if (!stopped) {
    if (tid < T_DIM) {
      const float* sl = slab + (size_t)((MAX_IT - 1) & 1) * (NBLK * T_DIM);
      float acc = 0.f;
      #pragma unroll 8
      for (int bb = 0; bb < NBLK; ++bb)
        acc += __hip_atomic_load(sl + bb * T_DIM + tid, __ATOMIC_RELAXED,
                                 __HIP_MEMORY_SCOPE_AGENT);
      u_lds[tid] = A0 / (acc + EPS_F);
    }
    __syncthreads();
  }

  // ---- loss = sum_t,j u_t K[t,j] v_j cost[t,j] -------------------------
  float acc = 0.f;
  if (jvalid) {
    const unsigned* colp = kpair + myj;
    const float* cc = cost + myj;
    #pragma unroll 4
    for (int t = 0; t < T_DIM; ++t) {
      unsigned kp = colp[t * V_DIM];
      float kv = bfhi_f32(kp) + bflo_f32(kp);
      acc = fmaf(kv * cc[t * V_DIM], u_lds[t], acc);
    }
    acc *= v_reg;
  }
  red[tid] = acc;
  __syncthreads();
  for (int s2 = TPB / 2; s2 > 0; s2 >>= 1) {
    if (tid < s2) red[tid] += red[tid + s2];
    __syncthreads();
  }
  if (tid == 0) partials[b] = red[0];
}

__global__ void sinkhorn_final(const float* __restrict__ partials,
                               float* __restrict__ out) {
  if (blockIdx.x == 0 && threadIdx.x == 0) {
    float s = 0.f;
    for (int bb = 0; bb < NBLK; ++bb) s += partials[bb];  // fixed order
    out[0] = s * W_LOSS;
  }
}

extern "C" void kernel_launch(void* const* d_in, const int* in_sizes, int n_in,
                              void* d_out, int out_size, void* d_ws, size_t ws_size,
                              hipStream_t stream) {
  (void)in_sizes; (void)n_in; (void)out_size; (void)ws_size;
  const float* cost = (const float*)d_in[0];

  char* ws = (char*)d_ws;
  size_t o = 0;
  const size_t nel = (size_t)T_DIM * V_DIM + PAD_EL;
  unsigned*       kpair = (unsigned*)(ws + o);       o += nel * 4;
  unsigned short* khp   = (unsigned short*)(ws + o); o += nel * 2;
  unsigned short* klp   = (unsigned short*)(ws + o); o += nel * 2;
  float*    slab  = (float*)(ws + o);    o += (size_t)2 * NBLK * T_DIM * 4;
  float*    errp  = (float*)(ws + o);    o += (size_t)16 * NBLK * 4;
  float*    parts = (float*)(ws + o);    o += (size_t)NBLK * 4;
  unsigned* bar   = (unsigned*)(ws + o); o += 16;
  // total ws use: ~61.5 MB (L2 is 192 MB)

  const int init_blocks = (int)((nel + 255) / 256);
  sinkhorn_init<<<init_blocks, 256, 0, stream>>>(cost, kpair, khp, klp, bar);
  sinkhorn_persist<<<NBLK, TPB, 0, stream>>>(cost, kpair, khp, klp, slab,
                                             errp, parts, bar);
  sinkhorn_final<<<1, 64, 0, stream>>>(parts, (float*)d_out);
}
